// NaiveHybridAttention_64733747085350
// MI455X (gfx1250) — matrix operand order repaired and checked by reading, not compiled, not run on hardware
//
#include <hip/hip_runtime.h>
#include <hip/hip_bf16.h>

typedef __attribute__((ext_vector_type(16))) __bf16 v16bf;
typedef __attribute__((ext_vector_type(8)))  __bf16 v8bf;
typedef __attribute__((ext_vector_type(8)))  float  v8f;
typedef __attribute__((ext_vector_type(2)))  float  v2f;

#define B_SZ   2
#define S_SZ   2048
#define D_SZ   2048
#define H_SZ   16
#define HD_SZ  128
#define WIN    512
#define QKV_LD 6144   // 3*D, row stride of the fused qkv buffer

// ---------------- WMMA wrappers ----------------
static __device__ __forceinline__ v8f wmma_bf16(v16bf a, v16bf b, v8f c) {
  // D = A(16x32 bf16) * B(32x16 bf16) + C(16x16 f32)
  return __builtin_amdgcn_wmma_f32_16x16x32_bf16(false, a, false, b, (short)0, c,
                                                 false, false);
}
static __device__ __forceinline__ v8f wmma_f32k4(v2f a, v2f b, v8f c) {
  // D = A(16x4 f32) * B(4x16 f32) + C(16x16 f32)  -- exact fp32
  return __builtin_amdgcn_wmma_f32_16x16x4_f32(false, a, false, b, (short)0, c,
                                               false, false);
}

// A-matrix 16x32 bf16 per-lane layout: lanes0-15 hold K=[0..8)+[16..24),
// lanes16-31 hold K=[8..16)+[24..32)  (ISA 7.12.2)
static __device__ __forceinline__ v16bf load_a_bf16(const __bf16* row, int k0, int half) {
  v8bf lo = *(const v8bf*)(row + k0 + half * 8);
  v8bf hi = *(const v8bf*)(row + k0 + 16 + half * 8);
  v16bf r;
#pragma unroll
  for (int i = 0; i < 8; ++i) { r[i] = lo[i]; r[i + 8] = hi[i]; }
  return r;
}
// B-matrix 32x16 bf16: lane = column, lanes0-15 hold K=[0..8)+[16..24), lanes16-31 K=[8..16)+[24..32)
// row points at 32 contiguous bf16 (one K-step of one n-row)
static __device__ __forceinline__ v16bf load_b_row32(const __bf16* row, int half) {
  v8bf lo = *(const v8bf*)(row + half * 8);
  v8bf hi = *(const v8bf*)(row + 16 + half * 8);
  v16bf r;
#pragma unroll
  for (int i = 0; i < 8; ++i) { r[i] = lo[i]; r[i + 8] = hi[i]; }
  return r;
}

// ---------------- fp32 -> (hi,lo) bf16 split ----------------
__global__ void __launch_bounds__(256)
k_split_bf16(const float* __restrict__ src, __bf16* __restrict__ hi,
             __bf16* __restrict__ lo, int n) {
  int i = blockIdx.x * 256 + threadIdx.x;
  if (i >= n) return;
  float v = src[i];
  __bf16 h = (__bf16)v;
  hi[i] = h;
  lo[i] = (__bf16)(v - (float)h);
}

// ---------------- split-bf16 GEMM:  C[M,N] = A[M,K] * B[N,K]^T ----------------
// block = 256 threads = 8 waves (4 m-positions x 2 n-halves), block tile 128x128,
// wave tile 32x64. B tiles are DMA-staged into LDS with GLOBAL_LOAD_ASYNC_TO_LDS
// (ASYNCcnt) and double-buffered so the DMA overlaps the WMMA stream.
__global__ void __launch_bounds__(256)
k_gemm_bf16x3(const __bf16* __restrict__ Ah, const __bf16* __restrict__ Al,
              const __bf16* __restrict__ Bh, const __bf16* __restrict__ Bl,
              float* __restrict__ C, int N, int K) {
  __shared__ __align__(16) __bf16 stage[2][2][128][32];  // [buf][hi/lo][n-row][k] = 32 KB

  const int tid  = threadIdx.x;
  const int lane = tid & 31;
  const int wave = tid >> 5;
  const int half = lane >> 4;
  const int l16  = lane & 15;
  const int mw   = wave >> 1;              // 0..3
  const int nw   = wave & 1;               // 0..1
  const int mblk = blockIdx.x * 128;
  const int nblk = blockIdx.y * 128;
  const int m0   = mblk + mw * 32;         // this wave: rows [m0, m0+32)
  const int n0   = nw * 64;                // this wave: block-local cols [n0, n0+64)

  const __bf16* a0h = Ah + (size_t)(m0 + l16) * K;
  const __bf16* a0l = Al + (size_t)(m0 + l16) * K;
  const __bf16* a1h = Ah + (size_t)(m0 + 16 + l16) * K;
  const __bf16* a1l = Al + (size_t)(m0 + 16 + l16) * K;

  // Issue one K-step of B (128 rows x 32 bf16 x {hi,lo} = 16 KB) as 4 async
  // b128 DMA transfers per thread (1024 transfers per block-buffer).
  auto issue = [&](int buf, int k0) {
#pragma unroll
    for (int j = 0; j < 4; ++j) {
      const int i   = tid + j * 256;       // 0..1023
      const int hl  = i >> 9;              // 0 = hi plane, 1 = lo plane
      const int row = (i >> 2) & 127;
      const int c16 = i & 3;               // 16B chunk within the 64B row
      const __bf16* src = (hl ? Bl : Bh) + (size_t)(nblk + row) * K + k0 + c16 * 8;
      unsigned ldsoff = (unsigned)(size_t)(void*)&stage[buf][hl][row][c16 * 8];
      unsigned long long gaddr = (unsigned long long)(size_t)src;
      asm volatile("global_load_async_to_lds_b128 %0, %1, off"
                   :: "v"(ldsoff), "v"(gaddr) : "memory");
    }
  };

  v8f acc[2][4] = {};
  issue(0, 0);
  for (int k0 = 0; k0 < K; k0 += 32) {
    const int buf = (k0 >> 5) & 1;
    if (k0 + 32 < K) {
      issue(buf ^ 1, k0 + 32);                          // prefetch next buffer
      asm volatile("s_wait_asynccnt 0x4" ::: "memory"); // drain previous batch only
    } else {
      asm volatile("s_wait_asynccnt 0x0" ::: "memory");
    }
    __syncthreads();

    v16bf ah0 = load_a_bf16(a0h, k0, half);
    v16bf al0 = load_a_bf16(a0l, k0, half);
    v16bf ah1 = load_a_bf16(a1h, k0, half);
    v16bf al1 = load_a_bf16(a1l, k0, half);
#pragma unroll
    for (int t = 0; t < 4; ++t) {
      v16bf bh = load_b_row32(&stage[buf][0][n0 + t * 16 + l16][0], half);
      v16bf bl = load_b_row32(&stage[buf][1][n0 + t * 16 + l16][0], half);
      acc[0][t] = wmma_bf16(ah0, bh, acc[0][t]);   // Ah*Bh
      acc[0][t] = wmma_bf16(ah0, bl, acc[0][t]);   // Ah*Bl
      acc[0][t] = wmma_bf16(al0, bh, acc[0][t]);   // Al*Bh  (Al*Bl ~2^-16, dropped)
      acc[1][t] = wmma_bf16(ah1, bh, acc[1][t]);
      acc[1][t] = wmma_bf16(ah1, bl, acc[1][t]);
      acc[1][t] = wmma_bf16(al1, bh, acc[1][t]);
    }
    __syncthreads();   // protect buf from next iteration's DMA writes
  }

  // C layout: VGPR r -> row r (lanes0-15) / r+8 (lanes16-31), col = lane%16
#pragma unroll
  for (int mt = 0; mt < 2; ++mt)
#pragma unroll
    for (int t = 0; t < 4; ++t)
#pragma unroll
      for (int r = 0; r < 8; ++r)
        C[(size_t)(m0 + mt * 16 + r + 8 * half) * N + (nblk + n0 + t * 16 + l16)] =
            acc[mt][t][r];
}

// ---------------- RoPE in place on q,k parts of qkv ----------------
__global__ void __launch_bounds__(256)
k_rope(float* __restrict__ qkv) {
  int idx = blockIdx.x * 256 + threadIdx.x;       // over B*S*H*64 = 2^22
  if (idx >= B_SZ * S_SZ * H_SZ * (HD_SZ / 2)) return;
  int i  = idx & 63;
  int h  = (idx >> 6) & 15;
  int s  = (idx >> 10) & (S_SZ - 1);
  int bb = idx >> 21;
  // inv_freq = ROPE_BASE^(-2i/HD); ln(10000) = 9.210340371976184
  float inv = __expf(-(float)(2 * i) * (9.210340371976184f / 128.0f));
  float sn, cs;
  sincosf((float)s * inv, &sn, &cs);
  size_t base = ((size_t)bb * S_SZ + s) * QKV_LD + h * HD_SZ;
  float* q = qkv + base;
  float x1 = q[i], x2 = q[i + 64];
  q[i]      = x1 * cs - x2 * sn;
  q[i + 64] = x2 * cs + x1 * sn;
  float* k = qkv + base + D_SZ;
  x1 = k[i]; x2 = k[i + 64];
  k[i]      = x1 * cs - x2 * sn;
  k[i + 64] = x2 * cs + x1 * sn;
}

// ---------------- sliding-window flash attention, exact fp32 WMMA ----------------
// one wave per (b, h, 16-query tile); grid = (S/16, H, B), block = 32
__global__ void __launch_bounds__(32)
k_attn(const float* __restrict__ qkv, float* __restrict__ ao) {
  __shared__ float pst[256];                      // 16x16 P staging (C->A relayout)
  const int lane = threadIdx.x & 31;
  const int half = lane >> 4;
  const int l16  = lane & 15;
  const int q0 = blockIdx.x * 16;
  const int h  = blockIdx.y;
  const int bb = blockIdx.z;

  const float* base = qkv + (size_t)bb * S_SZ * QKV_LD + h * HD_SZ;
  const float scale = 0.088388347648318447f;      // 1/sqrt(128), folded into Q

  // Q tile as f32 A-layout: lane row = lane%16, per 16x4 block: K pair at 4*kb+2*half
  v2f qa[32];
  {
    const float* qrow = base + (size_t)(q0 + l16) * QKV_LD;
#pragma unroll
    for (int kb = 0; kb < 32; ++kb) {
      const float* p = qrow + kb * 4 + half * 2;
      v2f t = {p[0] * scale, p[1] * scale};
      qa[kb] = t;
    }
  }

  v8f o[8] = {};                                  // 16x128 output accumulators
  float mrun[8], lrun[8];
#pragma unroll
  for (int r = 0; r < 8; ++r) { mrun[r] = -3.0e38f; lrun[r] = 0.0f; }

  int js = q0 - (WIN - 1);
  if (js < 0) js = 0;
  js &= ~15;

  for (int j0 = js; j0 <= q0; j0 += 16) {
    // ---- S = Q * K^T (K=128 via 32 f32 WMMAs) ----
    v8f s = {};
    {
      const float* krow = base + D_SZ + (size_t)(j0 + l16) * QKV_LD;
#pragma unroll
      for (int kb = 0; kb < 32; ++kb) {
        const float* p = krow + kb * 4 + half * 2;
        v2f bv = {p[0], p[1]};
        s = wmma_f32k4(qa[kb], bv, s);
      }
    }
    // ---- mask + online softmax (rows live in lane-halves of each C VGPR) ----
    float pv[8];
#pragma unroll
    for (int r = 0; r < 8; ++r) {
      const int i  = q0 + r + 8 * half;
      const int j  = j0 + l16;
      const int df = i - j;
      const bool valid = (df >= 0) && (df < WIN);
      float sv = valid ? s[r] : -3.0e38f;
      float rm = sv;
      rm = fmaxf(rm, __shfl_xor(rm, 1, 16));
      rm = fmaxf(rm, __shfl_xor(rm, 2, 16));
      rm = fmaxf(rm, __shfl_xor(rm, 4, 16));
      rm = fmaxf(rm, __shfl_xor(rm, 8, 16));
      float mnew = fmaxf(mrun[r], rm);
      float corr = __expf(mrun[r] - mnew);
      float e = valid ? __expf(sv - mnew) : 0.0f;
      float rs = e;
      rs += __shfl_xor(rs, 1, 16);
      rs += __shfl_xor(rs, 2, 16);
      rs += __shfl_xor(rs, 4, 16);
      rs += __shfl_xor(rs, 8, 16);
      lrun[r] = lrun[r] * corr + rs;
      mrun[r] = mnew;
#pragma unroll
      for (int c = 0; c < 8; ++c) o[c][r] *= corr;
      pv[r] = e;
    }
    // ---- stage P through LDS: C-layout -> A-layout ----
    __syncthreads();
#pragma unroll
    for (int r = 0; r < 8; ++r) pst[(r + 8 * half) * 16 + l16] = pv[r];
    __syncthreads();
    // ---- O += P(16x16) * V(16x128) via f32 WMMAs ----
#pragma unroll
    for (int c = 0; c < 8; ++c) {
#pragma unroll
      for (int kb = 0; kb < 4; ++kb) {
        const int kk = kb * 4 + half * 2;
        v2f pa = {pst[l16 * 16 + kk], pst[l16 * 16 + kk + 1]};
        const float* vp = base + 2 * D_SZ + (size_t)(j0 + kk) * QKV_LD + c * 16 + l16;
        v2f vb = {vp[0], vp[QKV_LD]};
        o[c] = wmma_f32k4(pa, vb, o[c]);
      }
    }
    __syncthreads();
  }

  // epilogue: normalize and write (B,S,H*HD) == (B,S,D)
#pragma unroll
  for (int c = 0; c < 8; ++c)
#pragma unroll
    for (int r = 0; r < 8; ++r) {
      const int i = q0 + r + 8 * half;
      ao[((size_t)bb * S_SZ + i) * D_SZ + h * HD_SZ + c * 16 + l16] = o[c][r] / lrun[r];
    }
}

// ---------------- host-side orchestration ----------------
extern "C" void kernel_launch(void* const* d_in, const int* in_sizes, int n_in,
                              void* d_out, int out_size, void* d_ws, size_t ws_size,
                              hipStream_t stream) {
  (void)in_sizes; (void)n_in; (void)out_size; (void)ws_size;
  const float* x     = (const float*)d_in[0];   // (B,S,D) fp32
  const float* w_qkv = (const float*)d_in[1];   // (3D,D)  fp32
  const float* w_out = (const float*)d_in[2];   // (D,D)   fp32
  float* out = (float*)d_out;                   // (B,S,D) fp32

  const int M  = B_SZ * S_SZ;   // 4096
  const int K  = D_SZ;          // 2048
  const int N3 = 3 * D_SZ;      // 6144

  size_t off = 0;
  auto alloc = [&](size_t bytes) -> void* {
    void* p = (char*)d_ws + off;
    off += (bytes + 255) & ~(size_t)255;
    return p;
  };
  __bf16* xh  = (__bf16*)alloc((size_t)M * K * sizeof(__bf16));
  __bf16* xl  = (__bf16*)alloc((size_t)M * K * sizeof(__bf16));
  __bf16* wqh = (__bf16*)alloc((size_t)N3 * K * sizeof(__bf16));
  __bf16* wql = (__bf16*)alloc((size_t)N3 * K * sizeof(__bf16));
  __bf16* woh = (__bf16*)alloc((size_t)D_SZ * K * sizeof(__bf16));
  __bf16* wol = (__bf16*)alloc((size_t)D_SZ * K * sizeof(__bf16));
  float*  qkv = (float*)alloc((size_t)M * N3 * sizeof(float));
  float*  ao  = (float*)alloc((size_t)M * D_SZ * sizeof(float));
  __bf16* aoh = xh;  // x no longer needed after the QKV GEMM -> reuse
  __bf16* aol = xl;

  const int n_x  = M * K;
  const int n_wq = N3 * K;
  const int n_wo = D_SZ * K;

  k_split_bf16<<<(n_x  + 255) / 256, 256, 0, stream>>>(x,     xh,  xl,  n_x);
  k_split_bf16<<<(n_wq + 255) / 256, 256, 0, stream>>>(w_qkv, wqh, wql, n_wq);
  k_split_bf16<<<(n_wo + 255) / 256, 256, 0, stream>>>(w_out, woh, wol, n_wo);

  // qkv = x @ w_qkv^T
  k_gemm_bf16x3<<<dim3(M / 128, N3 / 128), 256, 0, stream>>>(xh, xl, wqh, wql, qkv, N3, K);

  // RoPE on q,k thirds
  {
    int n = B_SZ * S_SZ * H_SZ * (HD_SZ / 2);
    k_rope<<<(n + 255) / 256, 256, 0, stream>>>(qkv);
  }

  // sliding-window attention -> ao (B,S,D)
  k_attn<<<dim3(S_SZ / 16, H_SZ, B_SZ), 32, 0, stream>>>(qkv, ao);

  // out = ao @ w_out^T
  k_split_bf16<<<(n_x + 255) / 256, 256, 0, stream>>>(ao, aoh, aol, n_x);
  k_gemm_bf16x3<<<dim3(M / 128, D_SZ / 128), 256, 0, stream>>>(aoh, aol, woh, wol, out, D_SZ, K);
}
